// localGNN_84542136254629
// MI455X (gfx1250) — matrix operand order repaired and checked by reading, not compiled, not run on hardware
//
#include <hip/hip_runtime.h>
#include <math.h>

// Problem dims (match reference)
#define G_    32
#define S_    128
#define NSUB  32
#define DEG   4
#define H     128
#define NSEG  (G_ * S_)        // 4096 subgraphs
#define EPS   (NSUB * DEG)     // 128 edges per subgraph

// Mapping: 1 workgroup = 4 waves (wave32) = 4 subgraphs
#define NSUBWG 4
#define WGSIZE (NSUBWG * 32)

// LDS row strides (halfs). 128 + 8 pad -> 272B rows, 16B aligned, bank-spread.
#define XSTR 136
#define WSTR 136

typedef _Float16 v16h __attribute__((ext_vector_type(16)));
typedef _Float16 v8h  __attribute__((ext_vector_type(8)));
typedef _Float16 v4h  __attribute__((ext_vector_type(4)));
typedef float    v8f  __attribute__((ext_vector_type(8)));

// Branch-free exact-GELU: erf via Abramowitz-Stegun 7.1.26 (|err| < 1.5e-7,
// far below f16 storage ulp). Uses hw-native v_rcp_f32 / v_exp_f32 only --
// no exec-mask divergence, ~15 VALU ops vs ~40+ branchy ops for ocml erff.
__device__ __forceinline__ float gelu_exact(float x) {
  const float ax = fabsf(x);
  const float t  = __builtin_amdgcn_rcpf(1.0f + 0.3275911f * ax);
  float p = 1.061405429f;
  p = p * t - 1.453152027f;
  p = p * t + 1.421413741f;
  p = p * t - 0.284496736f;
  p = p * t + 0.254829592f;
  p = p * t;
  // exp(-ax^2) = exp2(-log2(e) * ax^2)  -> single v_exp_f32
  const float e  = __builtin_amdgcn_exp2f(-1.4426950408889634f * ax * ax);
  const float er = copysignf(1.0f - p * e, x);
  return 0.5f * x * (1.0f + er);
}

// Cooperative: global f32 W[k][n] (row-major, k = input dim) -> LDS f16 Wt[n][k]
// (transposed so B-fragments read contiguous K).
__device__ __forceinline__ void load_weight_t(const float* __restrict__ W,
                                              _Float16* __restrict__ Wt, int tid) {
  for (int base = tid * 4; base < H * H; base += WGSIZE * 4) {
    int k = base >> 7;        // row of W (K index)
    int n = base & 127;       // starting column
    float4 v = *(const float4*)(W + base);
    Wt[(n + 0) * WSTR + k] = (_Float16)v.x;
    Wt[(n + 1) * WSTR + k] = (_Float16)v.y;
    Wt[(n + 2) * WSTR + k] = (_Float16)v.z;
    Wt[(n + 3) * WSTR + k] = (_Float16)v.w;
  }
}

// One 32x128 = (32x128)*(128x128) GEMM + bias + GELU, per wave.
// src: f16 [32][XSTR] activations, wt: f16 [128][WSTR] transposed weight,
// bias: f32[128] global, dst: f16 [32][XSTR].
__device__ __forceinline__ void gemm_gelu(const _Float16* src,
                                          const _Float16* wt,
                                          const float* __restrict__ bias,
                                          _Float16* dst, int lane) {
  const int mrow = lane & 15;   // A row / B,C column within tile
  const int hi   = lane >> 4;   // lane-half selector

  // Hoist the per-tile bias loads (one global f32 per nt) out of the WMMA loop.
  float bv[8];
#pragma unroll
  for (int nt = 0; nt < 8; ++nt) bv[nt] = bias[nt * 16 + mrow];

  // Preload all A fragments: 2 M-tiles x 4 K-steps (16x32 f16 each).
  // Per ISA layout: VGPR0-3 = K (hi*8 .. hi*8+7), VGPR4-7 = K+16.
  v16h A[2][4];
#pragma unroll
  for (int mt = 0; mt < 2; ++mt) {
#pragma unroll
    for (int k = 0; k < 4; ++k) {
      const _Float16* p = src + (mt * 16 + mrow) * XSTR + k * 32 + hi * 8;
      union { v16h v; v8h h[2]; } u;
      u.h[0] = *(const v8h*)(p);
      u.h[1] = *(const v8h*)(p + 16);
      A[mt][k] = u.v;
    }
  }

#pragma unroll
  for (int nt = 0; nt < 8; ++nt) {
    const int col = nt * 16 + mrow;
    // B fragments: lane = column, halfs K = hi*8 .. hi*8+7 and K+16 in Wt row.
    v16h B[4];
#pragma unroll
    for (int k = 0; k < 4; ++k) {
      const _Float16* bp = wt + col * WSTR + k * 32 + hi * 8;
      union { v16h v; v8h h[2]; } u;
      u.h[0] = *(const v8h*)(bp);
      u.h[1] = *(const v8h*)(bp + 16);
      B[k] = u.v;
    }
#pragma unroll
    for (int mt = 0; mt < 2; ++mt) {
      v8f acc = {0.f, 0.f, 0.f, 0.f, 0.f, 0.f, 0.f, 0.f};
#pragma unroll
      for (int k = 0; k < 4; ++k) {
        acc = __builtin_amdgcn_wmma_f32_16x16x32_f16(
            false, A[mt][k], false, B[k], (short)0, acc, false, false);
      }
      // C/D layout: lane -> column, VGPR r -> row (r + hi*8)
#pragma unroll
      for (int r = 0; r < 8; ++r) {
        const int m = mt * 16 + hi * 8 + r;
        dst[m * XSTR + col] = (_Float16)gelu_exact(acc[r] + bv[nt]);
      }
    }
  }
}

// GIN aggregation for one subgraph (per wave): agg = x + sum_{e: dst} x[src].
// Each lane owns 4 contiguous features (8B -> ds_*_b64, v_pk_add_f16).
__device__ __forceinline__ void aggregate(const _Float16* xb, _Float16* agg,
                                          const int* es, const int* ed, int lane) {
  const int fo = lane * 4;
#pragma unroll 4
  for (int i = 0; i < NSUB; ++i)
    *(v4h*)(agg + i * XSTR + fo) = *(const v4h*)(xb + i * XSTR + fo);
  for (int e = 0; e < EPS; ++e) {
    const int sl = es[e];
    const int dl = ed[e];
    v4h v = *(const v4h*)(xb + sl * XSTR + fo);
    v4h* ap = (v4h*)(agg + dl * XSTR + fo);
    *ap = *ap + v;   // in-order per lane; lanes own disjoint features
  }
}

__global__ void __launch_bounds__(WGSIZE)
gin_fused_kernel(const int* __restrict__ edge_src, const int* __restrict__ edge_dst,
                 const float* __restrict__ w0a, const float* __restrict__ b0a,
                 const float* __restrict__ w0b, const float* __restrict__ b0b,
                 const float* __restrict__ w1a, const float* __restrict__ b1a,
                 const float* __restrict__ w1b, const float* __restrict__ b1b,
                 const float* __restrict__ w2a, const float* __restrict__ b2a,
                 const float* __restrict__ w2b, const float* __restrict__ b2b,
                 const float* __restrict__ fc0_w, const float* __restrict__ fc0_b,
                 const float* __restrict__ fc1_w, const float* __restrict__ fc1_b,
                 float* __restrict__ out) {
  __shared__ __align__(16) _Float16 sWA[H * WSTR];            // 34 KB
  __shared__ __align__(16) _Float16 sWB[H * WSTR];            // 34 KB
  __shared__ __align__(16) _Float16 sX[NSUBWG][NSUB * XSTR];  // 4x8.5 KB
  __shared__ __align__(16) _Float16 sAG[NSUBWG][NSUB * XSTR];
  __shared__ __align__(16) _Float16 sHH[NSUBWG][NSUB * XSTR];
  __shared__ int   sES[NSUBWG][EPS];
  __shared__ int   sED[NSUBWG][EPS];
  __shared__ float sDEG[NSUBWG][NSUB];

  const int tid  = threadIdx.x;
  const int w    = tid >> 5;
  const int lane = tid & 31;
  const int sg   = blockIdx.x * NSUBWG + w;

  // ---- stage this wave's subgraph edges as local indices ----
  {
    const int eb = sg * EPS, nb = sg * NSUB;
    for (int t = lane; t < EPS; t += 32) {
      sES[w][t] = edge_src[eb + t] - nb;
      sED[w][t] = edge_dst[eb + t] - nb;
    }
  }
  // cooperative: layer-0 second weight (w0b) -> sWB
  load_weight_t(w0b, sWB, tid);

  // ---- in-degree (x = ones, so agg0[i] = 1 + indeg(i)) ----
  {
    int cnt = 0;
    for (int e = 0; e < EPS; ++e) cnt += (sED[w][e] == lane) ? 1 : 0;
    sDEG[w][lane] = 1.0f + (float)cnt;
  }

  // ---- layer-0 first half: h = gelu(agg0 * w0a + b0a)  (w0a is 1xH) ----
  {
    float wa[4], ba[4];
#pragma unroll
    for (int c = 0; c < 4; ++c) {
      wa[c] = w0a[c * 32 + lane];
      ba[c] = b0a[c * 32 + lane];
    }
    for (int i = 0; i < NSUB; ++i) {
      const float d = sDEG[w][i];
#pragma unroll
      for (int c = 0; c < 4; ++c)
        sHH[w][i * XSTR + c * 32 + lane] = (_Float16)gelu_exact(d * wa[c] + ba[c]);
    }
  }
  __syncthreads();
  gemm_gelu(sHH[w], sWB, b0b, sX[w], lane);   // x1 = gelu(h @ w0b + b0b)
  __syncthreads();

  // ---- GIN layers 1 and 2 ----
  for (int L = 1; L <= 2; ++L) {
    const float* wA = (L == 1) ? w1a : w2a;
    const float* bA = (L == 1) ? b1a : b2a;
    const float* wB = (L == 1) ? w1b : w2b;
    const float* bB = (L == 1) ? b1b : b2b;
    load_weight_t(wA, sWA, tid);
    load_weight_t(wB, sWB, tid);
    aggregate(sX[w], sAG[w], sES[w], sED[w], lane);
    __syncthreads();
    gemm_gelu(sAG[w], sWA, bA, sHH[w], lane);  // h = gelu(agg @ wa + ba)
    gemm_gelu(sHH[w], sWB, bB, sX[w], lane);   // x = gelu(h @ wb + bb)
    __syncthreads();
  }

  // ---- fc0 + GELU ----
  load_weight_t(fc0_w, sWA, tid);
  __syncthreads();
  gemm_gelu(sX[w], sWA, fc0_b, sHH[w], lane);

  // ---- global_add_pool per subgraph, then fc1 -> one scalar ----
  {
    float s = 0.f;
#pragma unroll
    for (int c = 0; c < 4; ++c) {
      const int f = c * 32 + lane;
      float col = 0.f;
      for (int i = 0; i < NSUB; ++i) col += (float)sHH[w][i * XSTR + f];
      s += col * fc1_w[f];
    }
#pragma unroll
    for (int off = 16; off > 0; off >>= 1) s += __shfl_down(s, off, 32);
    if (lane == 0) out[sg] = s + fc1_b[0];
  }
}

extern "C" void kernel_launch(void* const* d_in, const int* in_sizes, int n_in,
                              void* d_out, int out_size, void* d_ws, size_t ws_size,
                              hipStream_t stream) {
  const int*   edge_src = (const int*)d_in[0];
  const int*   edge_dst = (const int*)d_in[1];
  // d_in[2] (node2sub) unused: node i belongs to subgraph i / NSUB by construction
  const float* w0a   = (const float*)d_in[3];
  const float* b0a   = (const float*)d_in[4];
  const float* w0b   = (const float*)d_in[5];
  const float* b0b   = (const float*)d_in[6];
  const float* w1a   = (const float*)d_in[7];
  const float* b1a   = (const float*)d_in[8];
  const float* w1b   = (const float*)d_in[9];
  const float* b1b   = (const float*)d_in[10];
  const float* w2a   = (const float*)d_in[11];
  const float* b2a   = (const float*)d_in[12];
  const float* w2b   = (const float*)d_in[13];
  const float* b2b   = (const float*)d_in[14];
  const float* fc0_w = (const float*)d_in[15];
  const float* fc0_b = (const float*)d_in[16];
  const float* fc1_w = (const float*)d_in[17];
  const float* fc1_b = (const float*)d_in[18];
  float* out = (float*)d_out;

  dim3 grid(NSEG / NSUBWG);   // 1024 workgroups, one subgraph per wave
  dim3 block(WGSIZE);         // 128 threads = 4 wave32
  hipLaunchKernelGGL(gin_fused_kernel, grid, block, 0, stream,
                     edge_src, edge_dst,
                     w0a, b0a, w0b, b0b,
                     w1a, b1a, w1b, b1b,
                     w2a, b2a, w2b, b2b,
                     fc0_w, fc0_b, fc1_w, fc1_b, out);
}
